// AffineCL_71124658421940
// MI455X (gfx1250) — matrix operand order repaired and checked by reading, not compiled, not run on hardware
//
#include <hip/hip_runtime.h>

#define D_NODES 512
#define DM1     511
#define NH      24
#define BATCH   32768
#define SLOPE   0.2f

// ws blob layout per node (units of u32):
//   [net][ B2 tile0 (256) | B2 tile1 (256) | B3 tile0 (256) | B3 tile1 (256) | B4 tile (256) ] = 1280/net
//   small weights at 2560, 256 u32 per net, zero-padded to width 32:
//     [ W1(32) | b1(32) | b2(32) | b3(32) | b4(1) | pad ]
#define TILE_U32  256
#define NET_U32   1280
#define SMALL_OFF 2560
#define SMALL_NET 256
#define NODE_U32  3072   // 12 KB per node

typedef __attribute__((ext_vector_type(16))) __bf16        v16bf;
typedef __attribute__((ext_vector_type(8)))  float         v8f;
typedef __attribute__((ext_vector_type(4)))  unsigned int  v4u;
typedef __attribute__((ext_vector_type(8)))  int           v8i;
typedef __attribute__((ext_vector_type(4)))  int           v4i;

union ABits { unsigned int u[8]; v16bf v; };

static __device__ __forceinline__ unsigned short f2bf(float f) {
  unsigned int u = __builtin_bit_cast(unsigned int, f);
  u += 0x7FFFu + ((u >> 16) & 1u);           // RNE (used only in the pack kernel)
  return (unsigned short)(u >> 16);
}

// Pack two f32 -> {lo,hi} bf16 in one v_perm_b32 (truncation rounding).
static __device__ __forceinline__ unsigned int pk_bf16(float lo, float hi) {
  return __builtin_amdgcn_perm(__builtin_bit_cast(unsigned int, hi),
                               __builtin_bit_cast(unsigned int, lo),
                               0x07060302u);
}

// Branchless LeakyReLU for slope in (0,1): max(t, slope*t).
static __device__ __forceinline__ float lrelu(float t) {
  return fmaxf(t, SLOPE * t);
}

// ---------------------------------------------------------------------------
// TDM: async-DMA one 12 KB node blob (3072 dwords, 1-D tile) global -> LDS.
// D# built per CDNA5 ISA 8.3/8.4. Issued wave-uniform; tracked by TENSORcnt.
// ---------------------------------------------------------------------------
static __device__ __forceinline__ void tdm_load_node(const unsigned int* gsrc,
                                                     unsigned int lds_byte_off) {
  unsigned long long ga = (unsigned long long)(size_t)gsrc;
  v4u g0;
  g0[0] = 1u;                                        // count=1, user mode, no gather
  g0[1] = lds_byte_off;                              // lds_addr (bytes)
  g0[2] = (unsigned int)ga;                          // global_addr[31:0]
  g0[3] = (unsigned int)(ga >> 32) | 0x80000000u;    // global_addr[56:32] | type=2
  v8i g1;
  g1[0] = (int)(2u << 16);                           // data_size=4B; mask=0; no flags
  g1[1] = (int)((NODE_U32 & 0xFFFFu) << 16);         // tensor_dim0[15:0] @bits63:48
  g1[2] = (int)(((NODE_U32 >> 16) & 0xFFFFu) | (1u << 16)); // dim0[31:16] | tensor_dim1=1
  g1[3] = (int)((NODE_U32 & 0xFFFFu) << 16);         // tile_dim0=3072 @bits127:112
  g1[4] = 0;                                         // tile_dim1=0, tile_dim2=0
  g1[5] = (int)NODE_U32;                             // tensor_dim0_stride[31:0]
  g1[6] = 0;                                         // stride hi | dim1_stride lo
  g1[7] = 0;
  v4i z = {0, 0, 0, 0};
#if defined(__clang_major__) && __clang_major__ >= 23
  v8i z8 = {0, 0, 0, 0, 0, 0, 0, 0};
  __builtin_amdgcn_tensor_load_to_lds(g0, g1, z, z, z8, 0);
#else
  __builtin_amdgcn_tensor_load_to_lds(g0, g1, z, z, 0);
#endif
}

// ---------------------------------------------------------------------------
// Pack f32 weights -> bf16 WMMA B-operand layout (+ zero-padded small weights).
// One block per node.
// ---------------------------------------------------------------------------
__global__ __launch_bounds__(256) void pack_weights(
    const float* __restrict__ sW1, const float* __restrict__ sb1,
    const float* __restrict__ sW2, const float* __restrict__ sb2,
    const float* __restrict__ sW3, const float* __restrict__ sb3,
    const float* __restrict__ sW4, const float* __restrict__ sb4,
    const float* __restrict__ tW1, const float* __restrict__ tb1,
    const float* __restrict__ tW2, const float* __restrict__ tb2,
    const float* __restrict__ tW3, const float* __restrict__ tb3,
    const float* __restrict__ tW4, const float* __restrict__ tb4,
    unsigned int* __restrict__ ws) {
  const int node = blockIdx.x;                       // 0..510
  unsigned int* dst = ws + (size_t)node * NODE_U32;
  for (int j = 0; j < 12; ++j) {
    int idx = threadIdx.x + j * 256;
    if (idx < SMALL_OFF) {
      int net  = idx / NET_U32;
      int r    = idx % NET_U32;
      int tile = r / TILE_U32;                       // 0..4
      int e    = r % TILE_U32;
      int lane = e / 8, v = e % 8;
      // B-operand 32x16 bf16, this code's k order (same as A): lanes 0-15 hold K=0..15, lanes 16-31 K=16..31,
      // two K values per VGPR; N = lane % 16.
      int K0 = ((lane < 16) ? 0 : 16) + 2 * v;
      int n0 = lane & 15;
      float lo = 0.f, hi = 0.f;
      if (tile < 4) {
        const float* W = (tile < 2) ? (net ? tW2 : sW2) : (net ? tW3 : sW3);
        int N = (tile & 1) * 16 + n0;
        if (N < NH) {
          if (K0 < NH)     lo = W[(size_t)node * NH * NH + (size_t)K0 * NH + N];
          if (K0 + 1 < NH) hi = W[(size_t)node * NH * NH + (size_t)(K0 + 1) * NH + N];
        }
      } else {                                       // W4 as N=0 column tile
        const float* W4 = net ? tW4 : sW4;
        if (n0 == 0) {
          if (K0 < NH)     lo = W4[(size_t)node * NH + K0];
          if (K0 + 1 < NH) hi = W4[(size_t)node * NH + K0 + 1];
        }
      }
      dst[idx] = (unsigned int)f2bf(lo) | ((unsigned int)f2bf(hi) << 16);
    } else {
      // small weights, zero-padded so the chain kernel is branchless
      int q   = idx - SMALL_OFF;                     // 0..511
      int net = q / SMALL_NET, o = q % SMALL_NET;
      const float* W1 = net ? tW1 : sW1;
      const float* b1 = net ? tb1 : sb1;
      const float* b2 = net ? tb2 : sb2;
      const float* b3 = net ? tb3 : sb3;
      const float* b4 = net ? tb4 : sb4;
      float val = 0.f;
      if      (o < 32)  { int k = o;       if (k < NH) val = W1[(size_t)node * NH + k]; }
      else if (o < 64)  { int k = o - 32;  if (k < NH) val = b1[(size_t)node * NH + k]; }
      else if (o < 96)  { int k = o - 64;  if (k < NH) val = b2[(size_t)node * NH + k]; }
      else if (o < 128) { int k = o - 96;  if (k < NH) val = b3[(size_t)node * NH + k]; }
      else if (o == 128) val = b4[node];
      dst[idx] = __builtin_bit_cast(unsigned int, val);
    }
  }
}

// ---------------------------------------------------------------------------
// Chain kernel: each wave owns 16 batch rows, walks all 511 nodes.
// Block = 8 waves = 128 rows; grid = BATCH/128.
// Node weights double-buffered in LDS, prefetched by the Tensor Data Mover.
// ---------------------------------------------------------------------------
__global__ __launch_bounds__(256) void flow_chain(
    const float* __restrict__ x, const float* __restrict__ s0p,
    const float* __restrict__ t0p, const unsigned int* __restrict__ ws,
    float* __restrict__ out) {
  __shared__ unsigned int   blob[2][NODE_U32];       // double-buffered node weights
  __shared__ unsigned short hbuf[8][16 * 32];        // per-wave 16x32 bf16 transpose tile
  __shared__ float          resbuf[8][16];           // per-wave per-row MLP outputs

  const int tid  = threadIdx.x;
  const int wid  = tid >> 5;
  const int lane = tid & 31;
  const int half = lane >> 4;                        // K-group selector in A/B layouts
  const int rloc = lane & 15;                        // this lane's batch row within tile
  const int row  = blockIdx.x * 128 + wid * 16 + rloc;

  const unsigned int lds0 = (unsigned int)(size_t)&blob[0][0];
  const unsigned int lds1 = (unsigned int)(size_t)&blob[1][0];

  // Kick off TDM for node 0's weights while we do the root node.
  if (wid == 0) tdm_load_node(ws, lds0);

  // node 0 (root): z0 = exp(s0)*x0 + t0
  const float s0v = s0p[0], t0v = t0p[0];
  float z  = __expf(s0v) * x[(size_t)row * D_NODES] + t0v;
  float ld = s0v;
  if (half == 0) out[(size_t)row * D_NODES] = z;

  for (int i = 0; i < DM1; ++i) {
    if (wid == 0) __builtin_amdgcn_s_wait_tensorcnt(0);  // node i blob landed
    __syncthreads();                                     // publish; prev buffer free
    if (wid == 0 && (i + 1) < DM1)
      tdm_load_node(ws + (size_t)(i + 1) * NODE_U32, (i & 1) ? lds0 : lds1);

    const unsigned int* nb = &blob[i & 1][0];

    float res[2];
    #pragma unroll
    for (int net = 0; net < 2; ++net) {
      const unsigned int* wb = nb + net * NET_U32;
      const float* sw = (const float*)(nb + SMALL_OFF + net * SMALL_NET);

      // ---- layer 1: h1 = lrelu(z*W1 + b1), packed directly into A layout.
      // A 16x32 bf16, this code's k order (same as B): lane<16: V0-7 K=0..15;
      //                      lane>=16: V0-7 K=16..31.
      // W1/b1 zero-padded to 32 -> padded K lanes produce exact 0, branchless.
      ABits a;
      #pragma unroll
      for (int v = 0; v < 8; ++v) {
        int kb = 16 * half + 2 * v;
        float h0 = lrelu(z * sw[kb]     + sw[32 + kb]);
        float h1 = lrelu(z * sw[kb + 1] + sw[32 + kb + 1]);
        a.u[v] = pk_bf16(h0, h1);
      }

      // ---- layers 2 & 3: [16,32]x[32,32] via two 16x16x32 bf16 WMMAs each
      #pragma unroll
      for (int layer = 0; layer < 2; ++layer) {
        ABits b0, b1;
        #pragma unroll
        for (int v = 0; v < 8; ++v) {
          b0.u[v] = wb[(layer * 2 + 0) * TILE_U32 + lane * 8 + v];
          b1.u[v] = wb[(layer * 2 + 1) * TILE_U32 + lane * 8 + v];
        }
        v8f c0 = {}, c1 = {};
        c0 = __builtin_amdgcn_wmma_f32_16x16x32_bf16(false, a.v, false, b0.v,
                                                     (short)0, c0, false, false);
        c1 = __builtin_amdgcn_wmma_f32_16x16x32_bf16(false, a.v, false, b1.v,
                                                     (short)0, c1, false, false);
        const float* bias = sw + 64 + layer * 32;    // zero-padded to 32
        unsigned short* hb = &hbuf[wid][0];
        // D layout: lane holds column N=rloc (c0) / 16+rloc (c1), rows M=g+8*half.
        // Padded columns: c1==0, bias==0 -> lrelu(0)==0, no guard needed.
        #pragma unroll
        for (int g = 0; g < 8; ++g) {
          int M  = g + half * 8;
          float u0 = lrelu(c0[g] + bias[rloc]);
          float u1 = lrelu(c1[g] + bias[16 + rloc]);
          unsigned int p = pk_bf16(u0, u1);
          hb[M * 32 + rloc]      = (unsigned short)p;
          hb[M * 32 + 16 + rloc] = (unsigned short)(p >> 16);
        }
        __asm__ volatile("s_wait_dscnt 0" ::: "memory");   // intra-wave LDS transpose
        #pragma unroll
        for (int v = 0; v < 8; ++v) {
          int kb = 16 * half + 2 * v;
          a.u[v] = *(const unsigned int*)&hb[rloc * 32 + kb];
        }
      }

      // ---- layer 4: one WMMA against W4 column tile, result at N=0
      ABits b4t;
      #pragma unroll
      for (int v = 0; v < 8; ++v) b4t.u[v] = wb[4 * TILE_U32 + lane * 8 + v];
      v8f c = {};
      c = __builtin_amdgcn_wmma_f32_16x16x32_bf16(false, a.v, false, b4t.v,
                                                  (short)0, c, false, false);
      if (rloc == 0) {                               // lanes 0 (M=0..7) and 16 (M=8..15)
        float bb = sw[128];
        #pragma unroll
        for (int g = 0; g < 8; ++g) resbuf[wid][g + half * 8] = c[g] + bb;
      }
      __asm__ volatile("s_wait_dscnt 0" ::: "memory");
      res[net] = resbuf[wid][rloc];
    }

    const float xv = x[(size_t)row * D_NODES + (i + 1)];
    z   = __expf(res[0]) * xv + res[1];
    ld += res[0];
    if (half == 0) out[(size_t)row * D_NODES + (i + 1)] = z;
  }
  if (half == 0) out[(size_t)BATCH * D_NODES + row] = ld;   // log_det
}

// ---------------------------------------------------------------------------
extern "C" void kernel_launch(void* const* d_in, const int* in_sizes, int n_in,
                              void* d_out, int out_size, void* d_ws, size_t ws_size,
                              hipStream_t stream) {
  (void)in_sizes; (void)n_in; (void)out_size; (void)ws_size;
  const float* x   = (const float*)d_in[0];
  const float* s0  = (const float*)d_in[1];
  const float* t0  = (const float*)d_in[2];
  const float* sW1 = (const float*)d_in[3];
  const float* sb1 = (const float*)d_in[4];
  const float* sW2 = (const float*)d_in[5];
  const float* sb2 = (const float*)d_in[6];
  const float* sW3 = (const float*)d_in[7];
  const float* sb3 = (const float*)d_in[8];
  const float* sW4 = (const float*)d_in[9];
  const float* sb4 = (const float*)d_in[10];
  const float* tW1 = (const float*)d_in[11];
  const float* tb1 = (const float*)d_in[12];
  const float* tW2 = (const float*)d_in[13];
  const float* tb2 = (const float*)d_in[14];
  const float* tW3 = (const float*)d_in[15];
  const float* tb3 = (const float*)d_in[16];
  const float* tW4 = (const float*)d_in[17];
  const float* tb4 = (const float*)d_in[18];

  unsigned int* ws = (unsigned int*)d_ws;            // needs 511*3072*4 ≈ 6.3 MB

  pack_weights<<<DM1, 256, 0, stream>>>(sW1, sb1, sW2, sb2, sW3, sb3, sW4, sb4,
                                        tW1, tb1, tW2, tb2, tW3, tb3, tW4, tb4, ws);
  flow_chain<<<BATCH / 128, 256, 0, stream>>>(x, s0, t0, ws, (float*)d_out);
}
